// XLSTM_DM_51402168598682
// MI455X (gfx1250) — compile-verified, matrix operand order checked by reading, not run on hardware
//
#include <hip/hip_runtime.h>
#include <hip/hip_bf16.h>

// ---------------------------------------------------------------------------
// Problem constants (B=1)
// ---------------------------------------------------------------------------
#define SEQ    2048
#define DIM    1024
#define LAYERS 4
#define INNER  2048
#define NHEAD  4
#define DHEAD  512   // INNER / NHEAD
#define KCONV  4
#define UP2    4096  // 2*INNER
#define GIN    6144  // 3*INNER

typedef __attribute__((ext_vector_type(16))) _Float16 v16h;
typedef __attribute__((ext_vector_type(8)))  _Float16 v8h;
typedef __attribute__((ext_vector_type(8)))  float    v8f;

// ---------------------------------------------------------------------------
// WMMA helper (CDNA5: V_WMMA_F32_16X16X32_F16, f32 accumulate)
// ---------------------------------------------------------------------------
__device__ __forceinline__ v8f wmma_f16(v16h a, v16h b, v8f c) {
  return __builtin_amdgcn_wmma_f32_16x16x32_f16(
      /*neg_a=*/false, a, /*neg_b=*/false, b,
      /*c_mod=*/(short)0, c, /*reuse_a=*/false, /*reuse_b=*/false);
}

// Load a 16x32 f16 fragment from a row-major [rows, ld] matrix.
// ISA layout: lanes 0-15 -> row r, K {0..7,16..23}; lanes 16-31 -> row r, K {8..15,24..31}
__device__ __forceinline__ v16h load_frag(const _Float16* __restrict__ p, int ld) {
  const int lane = threadIdx.x & 31;
  const int r    = lane & 15;
  const int half = lane >> 4;
  const _Float16* q = p + (size_t)r * ld + half * 8;
  v8h lo = *reinterpret_cast<const v8h*>(q);
  v8h hi = *reinterpret_cast<const v8h*>(q + 16);
  v16h f;
#pragma unroll
  for (int i = 0; i < 8; ++i) { f[i] = lo[i]; f[i + 8] = hi[i]; }
  return f;
}

// ---------------------------------------------------------------------------
// CDNA5 async global->LDS copy (ASYNCcnt path, §10 async / 15.18.3)
// ---------------------------------------------------------------------------
__device__ __forceinline__ void async_ld_b128(unsigned lds_off, const _Float16* g) {
  asm volatile("global_load_async_to_lds_b128 %0, %1, off"
               :: "v"(lds_off), "v"(g) : "memory");
}
__device__ __forceinline__ void wait_async0() {
  asm volatile("s_wait_asynccnt 0x0" ::: "memory");
}
__device__ __forceinline__ unsigned lds_off32(const void* p) {
  // low 32 bits of a flat shared-aperture address == wave LDS byte offset
  return (unsigned)(unsigned long long)p;
}

// ---------------------------------------------------------------------------
// WMMA GEMM: C[M,N] = A[M,K] * B[N,K]^T (+bias[col]); optional C +=.
// Block tile 128x128, 8 waves each owning 64(M)x32(N) -> 8 v8f accumulators.
// A is double-buffered in LDS via async loads; B frags stream from global/L2.
// grid = (M/128, N/128), block = 256. K % 32 == 0, M % 128 == 0, N % 128 == 0.
// ---------------------------------------------------------------------------
#define APAD 40   // 128x32 f16 chunk, rows padded to 40 halves (80B): 16B-aligned,
                  // avoids the 256B-stride full bank collision on ds_load_b128
__global__ __launch_bounds__(256) void k_gemm_f16(
    const _Float16* __restrict__ A, const _Float16* __restrict__ B,
    const float* __restrict__ bias, float* __restrict__ C,
    int M, int N, int K, int ldc, int accumulate) {
  const int wave = threadIdx.x >> 5, lane = threadIdx.x & 31;
  const int col  = lane & 15,        half = lane >> 4;
  const int wm = wave >> 2;                 // 0..1  : M offset wm*64
  const int wn = wave & 3;                  // 0..3  : N offset wn*32
  const int mblk = blockIdx.x * 128;
  const int nblk = blockIdx.y * 128;
  const int m0 = mblk + wm * 64;
  const int n0 = nblk + wn * 32;

  __shared__ _Float16 As[2][128][APAD];     // 2 x 10KB staged A chunks

  v8f acc[4][2] = {};

  // stage K-chunk kk of A[128 rows] into buffer b (2 async b128 per thread)
  auto stage = [&](int b, int kk) {
#pragma unroll
    for (int i = 0; i < 2; ++i) {
      const int idx = threadIdx.x + i * 256;   // 0..511
      const int r = idx >> 2, c = idx & 3;     // row 0..127, 16B chunk 0..3
      async_ld_b128(lds_off32(&As[b][r][c * 8]),
                    A + (size_t)(mblk + r) * K + kk + c * 8);
    }
  };

  stage(0, 0);
  int b = 0;
  for (int kk = 0; kk < K; kk += 32, b ^= 1) {
    wait_async0();          // this wave's staged chunk landed
    __syncthreads();        // every wave's slice visible / prev reads done
    if (kk + 32 < K) stage(b ^ 1, kk + 32);
    if (kk + 64 < K)        // keep next B tiles warm in L2
      __builtin_prefetch(B + (size_t)n0 * K + kk + 64, 0, 1);

    v16h bf[2];
#pragma unroll
    for (int ni = 0; ni < 2; ++ni)
      bf[ni] = load_frag(B + (size_t)(n0 + ni * 16) * K + kk, K);
#pragma unroll
    for (int mi = 0; mi < 4; ++mi) {
      v16h af = load_frag(&As[b][wm * 64 + mi * 16][0], APAD);   // ds_load_b128
#pragma unroll
      for (int ni = 0; ni < 2; ++ni)
        acc[mi][ni] = wmma_f16(af, bf[ni], acc[mi][ni]);
    }
  }

#pragma unroll
  for (int ni = 0; ni < 2; ++ni) {
    const int c = n0 + ni * 16 + col;
    const float bv = bias ? bias[c] : 0.f;
#pragma unroll
    for (int mi = 0; mi < 4; ++mi) {
#pragma unroll
      for (int i = 0; i < 8; ++i) {
        const int r = m0 + mi * 16 + half * 8 + i;
        float v = acc[mi][ni][i] + bv;
        if (accumulate) C[(size_t)r * ldc + c] += v;
        else            C[(size_t)r * ldc + c]  = v;
      }
    }
  }
}

// ---------------------------------------------------------------------------
// LayerNorm over last dim (biased var, eps 1e-5), output f16 (feeds WMMA GEMMs)
// ---------------------------------------------------------------------------
__global__ __launch_bounds__(256) void k_layernorm(
    const float* __restrict__ x, const float* __restrict__ w,
    _Float16* __restrict__ y16, int D_) {
  const int row = blockIdx.x;
  const float* xr = x + (size_t)row * D_;
  __shared__ float s1[256], s2[256];
  float a = 0.f, b = 0.f;
  for (int c = threadIdx.x; c < D_; c += 256) { float v = xr[c]; a += v; b += v * v; }
  s1[threadIdx.x] = a; s2[threadIdx.x] = b;
  __syncthreads();
  for (int off = 128; off; off >>= 1) {
    if (threadIdx.x < off) { s1[threadIdx.x] += s1[threadIdx.x + off];
                             s2[threadIdx.x] += s2[threadIdx.x + off]; }
    __syncthreads();
  }
  const float mu  = s1[0] / D_;
  const float var = s2[0] / D_ - mu * mu;
  const float rs  = rsqrtf(var + 1e-5f);
  for (int c = threadIdx.x; c < D_; c += 256)
    y16[(size_t)row * D_ + c] = (_Float16)((xr[c] - mu) * rs * w[c]);
}

// ---------------------------------------------------------------------------
// Causal depthwise conv (K=4) + SiLU on xm = up[:, :INNER]
// ---------------------------------------------------------------------------
__global__ void k_conv_silu(const float* __restrict__ up, const float* __restrict__ cw,
                            const float* __restrict__ cb, float* __restrict__ xc) {
  const int idx = blockIdx.x * blockDim.x + threadIdx.x;
  if (idx >= SEQ * INNER) return;
  const int c = idx % INNER, s = idx / INNER;
  float acc = cb[c];
#pragma unroll
  for (int k = 0; k < KCONV; ++k) {
    const int t = s + k - (KCONV - 1);
    if (t >= 0) acc += up[(size_t)t * UP2 + c] * cw[c * KCONV + k];
  }
  xc[idx] = acc / (1.f + __expf(-acc));   // SiLU
}

// ---------------------------------------------------------------------------
// Block-diagonal 4x4 projections q,k (from xc) and v (from xm); emit f32 for
// gates, f16 q/k row-major, f16 v transposed [INNER, SEQ] for the W@V WMMA.
// ---------------------------------------------------------------------------
__global__ void k_headwise(const float* __restrict__ xc, const float* __restrict__ xm,
                           const float* __restrict__ qw, const float* __restrict__ kw,
                           const float* __restrict__ vw,
                           float* __restrict__ qf, float* __restrict__ kf,
                           float* __restrict__ vf,
                           _Float16* __restrict__ q16, _Float16* __restrict__ k16,
                           _Float16* __restrict__ vT16) {
  const int idx = blockIdx.x * blockDim.x + threadIdx.x;
  if (idx >= SEQ * INNER) return;
  const int c = idx % INNER, s = idx / INNER;
  const int n = c >> 2, o = c & 3;
  const float* wq = qw + ((n * 4 + o) * 4);
  const float* wk = kw + ((n * 4 + o) * 4);
  const float* wv = vw + ((n * 4 + o) * 4);
  float aq = 0.f, ak = 0.f, av = 0.f;
#pragma unroll
  for (int d = 0; d < 4; ++d) {
    const float xcv = xc[(size_t)s * INNER + n * 4 + d];
    const float xmv = xm[(size_t)s * UP2 + n * 4 + d];
    aq += xcv * wq[d]; ak += xcv * wk[d]; av += xmv * wv[d];
  }
  qf[idx] = aq; kf[idx] = ak; vf[idx] = av;
  q16[idx] = (_Float16)aq; k16[idx] = (_Float16)ak;
  vT16[(size_t)c * SEQ + s] = (_Float16)av;
}

// ---------------------------------------------------------------------------
// Gate dots: 8 waves per row s, wave j -> (head j&3, fg if j>=4), dot over 6144
// ---------------------------------------------------------------------------
__global__ __launch_bounds__(256) void k_gates(
    const float* __restrict__ qf, const float* __restrict__ kf,
    const float* __restrict__ vf,
    const float* __restrict__ igw, const float* __restrict__ igb,
    const float* __restrict__ fgw, const float* __restrict__ fgb,
    float* __restrict__ ig, float* __restrict__ fg) {
  const int s = blockIdx.x;
  const int wave = threadIdx.x >> 5, lane = threadIdx.x & 31;
  const int h = wave & 3, isfg = wave >> 2;
  const float* w = (isfg ? fgw : igw) + (size_t)h * GIN;
  float acc = 0.f;
  for (int e = lane; e < GIN; e += 32) {
    float g = (e < INNER) ? qf[(size_t)s * INNER + e]
            : (e < 2 * INNER ? kf[(size_t)s * INNER + e - INNER]
                             : vf[(size_t)s * INNER + e - 2 * INNER]);
    acc += g * w[e];
  }
  for (int off = 16; off; off >>= 1) acc += __shfl_xor(acc, off, 32);
  if (lane == 0) {
    const float b = (isfg ? fgb : igb)[h];
    (isfg ? fg : ig)[(size_t)h * SEQ + s] = acc + b;
  }
}

// ---------------------------------------------------------------------------
// Per-head O(S) scan: cs = cumsum(logsigmoid(fg)); a[t]=ig[t]-cs[t];
// M[t]=prefixmax(a); nfl[t]=exp(-cs[t]-M[t])   (the analytic stabilizer)
// ---------------------------------------------------------------------------
__global__ void k_scan(const float* __restrict__ ig, const float* __restrict__ fg,
                       float* __restrict__ aA, float* __restrict__ Ms,
                       float* __restrict__ nfl) {
  const int h = threadIdx.x;
  if (h >= NHEAD) return;
  float c = 0.f, mx = -3.4e38f;
  for (int t = 0; t < SEQ; ++t) {
    const float f  = fg[(size_t)h * SEQ + t];
    const float ls = fminf(f, 0.f) - log1pf(__expf(-fabsf(f)));  // log sigmoid
    c += ls;
    const float a = ig[(size_t)h * SEQ + t] - c;
    mx = fmaxf(mx, a);
    aA[(size_t)h * SEQ + t]  = a;
    Ms[(size_t)h * SEQ + t]  = mx;
    nfl[(size_t)h * SEQ + t] = __expf(-c - mx);
  }
}

// ---------------------------------------------------------------------------
// Fused mLSTM attention. grid = (SEQ/16, NHEAD), block = 256 (8 waves).
// Single pass over key tiles: per group of 8 tiles, each wave builds one
// decayed 16x16 C tile via WMMA (K=DHEAD), stages it f16 in LDS, reduces
// rowsums; then every wave runs W@V WMMAs for its 64-column DH slice.
// Final per-row scale 1/(max(|sum|, exp(-m)) + 1e-6).
// ---------------------------------------------------------------------------
#define CPAD 136   // 128 staged key cols padded to 136 halves (272B rows)
__global__ __launch_bounds__(256) void k_mlstm_attn(
    const _Float16* __restrict__ q16, const _Float16* __restrict__ k16,
    const _Float16* __restrict__ vT16,
    const float* __restrict__ aA, const float* __restrict__ Ms,
    const float* __restrict__ nfl, float* __restrict__ hOut) {
  const int qt = blockIdx.x, head = blockIdx.y;
  const int s0 = qt * 16;
  const int wave = threadIdx.x >> 5, lane = threadIdx.x & 31;
  const int col = lane & 15, half = lane >> 4;

  __shared__ _Float16 cbuf[16][CPAD];   // 8 staged 16x16 C tiles (f16)
  __shared__ float rowsum[16];
  __shared__ float rowscale[16];
  if (threadIdx.x < 16) rowsum[threadIdx.x] = 0.f;
  __syncthreads();

  float msv[8];
#pragma unroll
  for (int i = 0; i < 8; ++i) msv[i] = Ms[(size_t)head * SEQ + s0 + half * 8 + i];

  v8f Hacc[4] = {};                    // 64 DH columns per wave
  const float qscale = 0.04419417382415922f;   // 1/sqrt(512)
  const int ntiles = qt + 1;
  const int cbase  = head * DHEAD;

  for (int g = 0; g < ntiles; g += 8) {
    const int kt = g + wave;
    if (kt < ntiles) {
      const int t0 = kt * 16;
      v8f acc = {};
      for (int kk = 0; kk < DHEAD; kk += 32) {
        v16h af = load_frag(q16 + (size_t)s0 * INNER + cbase + kk, INNER);
        v16h bf = load_frag(k16 + (size_t)t0 * INNER + cbase + kk, INNER);
        acc = wmma_f16(af, bf, acc);
      }
      const int tcol = t0 + col;
      const float at = aA[(size_t)head * SEQ + tcol];
#pragma unroll
      for (int i = 0; i < 8; ++i) {
        const int srow = s0 + half * 8 + i;
        float cv = 0.f;
        if (tcol <= srow) cv = acc[i] * qscale * __expf(at - msv[i]);
        cbuf[half * 8 + i][wave * 16 + col] = (_Float16)cv;
        float v = cv;                              // rowsum across 16 lanes/half
        for (int off = 8; off; off >>= 1) v += __shfl_xor(v, off, 16);
        if (col == 0) atomicAdd(&rowsum[half * 8 + i], v);
      }
    } else {
#pragma unroll
      for (int i = 0; i < 8; ++i) cbuf[half * 8 + i][wave * 16 + col] = (_Float16)0.f;
    }
    __syncthreads();

    const int tbase  = g * 16;
    const int nvalid = (ntiles - g < 8 ? ntiles - g : 8) * 16;
#pragma unroll
    for (int p = 0; p < 4; ++p) {                  // 32 keys per WMMA K
      if (p * 32 < nvalid) {
        v16h af = load_frag(&cbuf[0][0] + p * 32, CPAD);   // ds_load path
#pragma unroll
        for (int nt = 0; nt < 4; ++nt) {
          const int c0 = cbase + wave * 64 + nt * 16;
          v16h bf = load_frag(vT16 + (size_t)c0 * SEQ + tbase + p * 32, SEQ);
          Hacc[nt] = wmma_f16(af, bf, Hacc[nt]);
        }
      }
    }
    __syncthreads();
  }

  if (threadIdx.x < 16) {
    const float n = fmaxf(fabsf(rowsum[threadIdx.x]),
                          nfl[(size_t)head * SEQ + s0 + threadIdx.x]);
    rowscale[threadIdx.x] = 1.f / (n + 1e-6f);
  }
  __syncthreads();
#pragma unroll
  for (int nt = 0; nt < 4; ++nt) {
    const int c0 = cbase + wave * 64 + nt * 16 + col;
#pragma unroll
    for (int i = 0; i < 8; ++i) {
      const int srow = s0 + half * 8 + i;
      hOut[(size_t)srow * INNER + c0] = Hacc[nt][i] * rowscale[half * 8 + i];
    }
  }
}

// ---------------------------------------------------------------------------
// Per-head LayerNorm over DH + skip*xc + silu(z) gating; f32 + f16 out
// ---------------------------------------------------------------------------
__global__ __launch_bounds__(256) void k_outnorm(
    const float* __restrict__ hIn, const float* __restrict__ xc,
    const float* __restrict__ up, const float* __restrict__ onw,
    const float* __restrict__ skip, float* __restrict__ outF,
    _Float16* __restrict__ out16) {
  const int s = blockIdx.x;
  const int h = threadIdx.x >> 6;   // head (4 x 64 threads)
  const int t = threadIdx.x & 63;
  __shared__ float r1[4][64], r2[4][64];
  const float* hr = hIn + (size_t)s * INNER + h * DHEAD;
  float a = 0.f, b = 0.f;
  for (int c = t; c < DHEAD; c += 64) { float v = hr[c]; a += v; b += v * v; }
  r1[h][t] = a; r2[h][t] = b;
  __syncthreads();
  for (int off = 32; off; off >>= 1) {
    if (t < off) { r1[h][t] += r1[h][t + off]; r2[h][t] += r2[h][t + off]; }
    __syncthreads();
  }
  const float mu   = r1[h][0] / DHEAD;
  const float var  = r2[h][0] / DHEAD - mu * mu;
  const float rstd = rsqrtf(var + 1e-5f);
  for (int c = t; c < DHEAD; c += 64) {
    const int cg = h * DHEAD + c;
    float v = (hr[c] - mu) * rstd * onw[cg] + skip[cg] * xc[(size_t)s * INNER + cg];
    const float z = up[(size_t)s * UP2 + INNER + cg];
    v *= z / (1.f + __expf(-z));
    outF[(size_t)s * INNER + cg]  = v;
    out16[(size_t)s * INNER + cg] = (_Float16)v;
  }
}

// ---------------------------------------------------------------------------
// Row dot for reward/term heads: out[s] = X[s,:]·w + b
// ---------------------------------------------------------------------------
__global__ void k_rowdot(const float* __restrict__ X, const float* __restrict__ w,
                         const float* __restrict__ b, float* __restrict__ out) {
  const int s = blockIdx.x, lane = threadIdx.x;
  float acc = 0.f;
  for (int c = lane; c < DIM; c += 32) acc += X[(size_t)s * DIM + c] * w[c];
  for (int off = 16; off; off >>= 1) acc += __shfl_xor(acc, off, 32);
  if (lane == 0) out[s] = acc + b[0];
}

__global__ void k_cvt_f16(const float* __restrict__ in, _Float16* __restrict__ out,
                          size_t n) {
  const size_t i = (size_t)blockIdx.x * blockDim.x + threadIdx.x;
  if (i < n) out[i] = (_Float16)in[i];
}

// ---------------------------------------------------------------------------
// Host orchestration
// ---------------------------------------------------------------------------
static inline size_t al256(size_t x) { return (x + 255) & ~(size_t)255; }

extern "C" void kernel_launch(void* const* d_in, const int* in_sizes, int n_in,
                              void* d_out, int out_size, void* d_ws, size_t ws_size,
                              hipStream_t stream) {
  const float* tokens = (const float*)d_in[0];
  const float* ln_w   = (const float*)d_in[1];
  const float* up_w   = (const float*)d_in[2];
  const float* conv_w = (const float*)d_in[3];
  const float* conv_b = (const float*)d_in[4];
  const float* q_w    = (const float*)d_in[5];
  const float* k_w    = (const float*)d_in[6];
  const float* v_w    = (const float*)d_in[7];
  const float* ig_w   = (const float*)d_in[8];
  const float* ig_b   = (const float*)d_in[9];
  const float* fg_w   = (const float*)d_in[10];
  const float* fg_b   = (const float*)d_in[11];
  const float* skip   = (const float*)d_in[12];
  const float* onw    = (const float*)d_in[13];
  const float* down_w = (const float*)d_in[14];
  const float* post_w = (const float*)d_in[15];
  const float* lat_w  = (const float*)d_in[16];
  const float* lat_b  = (const float*)d_in[17];
  const float* r1_w   = (const float*)d_in[18];
  const float* r1_b   = (const float*)d_in[19];
  const float* r2_w   = (const float*)d_in[20];
  const float* r2_b   = (const float*)d_in[21];
  const float* t1_w   = (const float*)d_in[22];
  const float* t1_b   = (const float*)d_in[23];
  const float* t2_w   = (const float*)d_in[24];
  const float* t2_b   = (const float*)d_in[25];
  float* dout = (float*)d_out;

  // ---- workspace carve-up ----
  char* p = (char*)d_ws;
  size_t o = 0;
  auto take = [&](size_t bytes) { void* r = p + o; o = al256(o + bytes); return r; };
  float*     x      = (float*)    take((size_t)SEQ * DIM * 4);
  _Float16*  xn16   = (_Float16*) take((size_t)SEQ * DIM * 2);
  float*     up     = (float*)    take((size_t)SEQ * UP2 * 4);
  float*     xc     = (float*)    take((size_t)SEQ * INNER * 4);
  float*     qf     = (float*)    take((size_t)SEQ * INNER * 4);
  float*     kf     = (float*)    take((size_t)SEQ * INNER * 4);
  float*     vf     = (float*)    take((size_t)SEQ * INNER * 4);
  _Float16*  q16    = (_Float16*) take((size_t)SEQ * INNER * 2);
  _Float16*  k16    = (_Float16*) take((size_t)SEQ * INNER * 2);
  _Float16*  vT16   = (_Float16*) take((size_t)INNER * SEQ * 2);
  float*     igB    = (float*)    take((size_t)NHEAD * SEQ * 4);
  float*     fgB    = (float*)    take((size_t)NHEAD * SEQ * 4);
  float*     aA     = (float*)    take((size_t)NHEAD * SEQ * 4);
  float*     MsB    = (float*)    take((size_t)NHEAD * SEQ * 4);
  float*     nflB   = (float*)    take((size_t)NHEAD * SEQ * 4);
  float*     hbuf   = (float*)    take((size_t)SEQ * INNER * 4);
  float*     outF   = (float*)    take((size_t)SEQ * INNER * 4);
  _Float16*  out16  = (_Float16*) take((size_t)SEQ * INNER * 2);
  float*     r1buf  = (float*)    take((size_t)SEQ * DIM * 4);
  _Float16*  upw16  = (_Float16*) take((size_t)LAYERS * UP2 * DIM * 2);
  _Float16*  dnw16  = (_Float16*) take((size_t)LAYERS * DIM * INNER * 2);
  _Float16*  latw16 = (_Float16*) take((size_t)DIM * DIM * 2);
  _Float16*  r1w16  = (_Float16*) take((size_t)DIM * DIM * 2);
  _Float16*  t1w16  = (_Float16*) take((size_t)DIM * DIM * 2);
  (void)ws_size; (void)n_in; (void)in_sizes; (void)out_size;

  // ---- one-time weight conversions to f16 ----
  auto cvt = [&](const float* src, _Float16* dst, size_t n) {
    k_cvt_f16<<<(unsigned)((n + 255) / 256), 256, 0, stream>>>(src, dst, n);
  };
  cvt(up_w,   upw16,  (size_t)LAYERS * UP2 * DIM);
  cvt(down_w, dnw16,  (size_t)LAYERS * DIM * INNER);
  cvt(lat_w,  latw16, (size_t)DIM * DIM);
  cvt(r1_w,   r1w16,  (size_t)DIM * DIM);
  cvt(t1_w,   t1w16,  (size_t)DIM * DIM);

  // x = tokens (mutable residual stream)
  hipMemcpyAsync(x, tokens, (size_t)SEQ * DIM * 4, hipMemcpyDeviceToDevice, stream);

  const int EW = SEQ * INNER / 256;   // elementwise grid

  for (int l = 0; l < LAYERS; ++l) {
    k_layernorm<<<SEQ, 256, 0, stream>>>(x, ln_w + (size_t)l * DIM, xn16, DIM);
    k_gemm_f16<<<dim3(SEQ / 128, UP2 / 128), 256, 0, stream>>>(
        xn16, upw16 + (size_t)l * UP2 * DIM, nullptr, up, SEQ, UP2, DIM, UP2, 0);
    k_conv_silu<<<EW, 256, 0, stream>>>(
        up, conv_w + (size_t)l * INNER * KCONV, conv_b + (size_t)l * INNER, xc);
    k_headwise<<<EW, 256, 0, stream>>>(
        xc, up,
        q_w + (size_t)l * INNER * 4, k_w + (size_t)l * INNER * 4,
        v_w + (size_t)l * INNER * 4,
        qf, kf, vf, q16, k16, vT16);
    k_gates<<<SEQ, 256, 0, stream>>>(
        qf, kf, vf,
        ig_w + (size_t)l * NHEAD * GIN, ig_b + (size_t)l * NHEAD,
        fg_w + (size_t)l * NHEAD * GIN, fg_b + (size_t)l * NHEAD, igB, fgB);
    k_scan<<<1, 32, 0, stream>>>(igB, fgB, aA, MsB, nflB);
    k_mlstm_attn<<<dim3(SEQ / 16, NHEAD), 256, 0, stream>>>(
        q16, k16, vT16, aA, MsB, nflB, hbuf);
    k_outnorm<<<SEQ, 256, 0, stream>>>(
        hbuf, xc, up, onw + (size_t)l * INNER, skip + (size_t)l * INNER, outF, out16);
    k_gemm_f16<<<dim3(SEQ / 128, DIM / 128), 256, 0, stream>>>(
        out16, dnw16 + (size_t)l * DIM * INNER, nullptr, x, SEQ, DIM, INNER, DIM, 1);
  }

  // ---- heads ----
  k_layernorm<<<SEQ, 256, 0, stream>>>(x, post_w, xn16, DIM);
  // latent -> d_out[0 : S*D)
  k_gemm_f16<<<dim3(SEQ / 128, DIM / 128), 256, 0, stream>>>(
      xn16, latw16, lat_b, dout, SEQ, DIM, DIM, DIM, 0);
  // reward -> d_out[S*D : S*D + S)
  k_gemm_f16<<<dim3(SEQ / 128, DIM / 128), 256, 0, stream>>>(
      xn16, r1w16, r1_b, r1buf, SEQ, DIM, DIM, DIM, 0);
  k_rowdot<<<SEQ, 32, 0, stream>>>(r1buf, r2_w, r2_b, dout + (size_t)SEQ * DIM);
  // term -> d_out[S*D + S : S*D + 2S)
  k_gemm_f16<<<dim3(SEQ / 128, DIM / 128), 256, 0, stream>>>(
      xn16, t1w16, t1_b, r1buf, SEQ, DIM, DIM, DIM, 0);
  k_rowdot<<<SEQ, 32, 0, stream>>>(r1buf, t2_w, t2_b, dout + (size_t)SEQ * DIM + SEQ);
}
